// Attention_85212151153298
// MI455X (gfx1250) — matrix operand order repaired and checked by reading, not compiled, not run on hardware
//
#include <hip/hip_runtime.h>

typedef __attribute__((ext_vector_type(16))) __bf16 v16bf;
typedef __attribute__((ext_vector_type(8)))  __bf16 v8bf;
typedef __attribute__((ext_vector_type(8)))  float  v8f;
typedef __attribute__((ext_vector_type(4)))  float  f32x4;
typedef __attribute__((ext_vector_type(4)))  int    v4i;

#define DIMX 1025
#define CTXX 4096
#define DPAD 1056     // 1025 padded to a multiple of 32 (WMMA K) / 16 (tiles)
#define DPAD128 1152  // DPAD padded to the 128-row tile grid (guard-free loads)
#define LMBD 0.9f

#if defined(__has_builtin)
#if __has_builtin(__builtin_amdgcn_global_load_async_to_lds_b128)
#define HAVE_ASYNC 1
#endif
#endif
#ifndef HAVE_ASYNC
#define HAVE_ASYNC 0
#endif

#if defined(__has_builtin)
#if __has_builtin(__builtin_amdgcn_s_wait_asynccnt)
#define WAIT_ASYNC(n) __builtin_amdgcn_s_wait_asynccnt(n)
#endif
#endif
#ifndef WAIT_ASYNC
#define WAIT_ASYNC(n) asm volatile("s_wait_asynccnt %0" ::"n"(n) : "memory")
#endif

// Builtin param types (from clang diagnostics): (v4i AS1*, v4i AS3*, Imm, Imm)
typedef v4i __attribute__((address_space(1)))* gv4i_ptr;
typedef v4i __attribute__((address_space(3)))* lv4i_ptr;
__device__ __forceinline__ gv4i_ptr to_gas(const void* p) {
  return (gv4i_ptr)(unsigned long long)(uintptr_t)p;  // global: value-preserving
}
__device__ __forceinline__ lv4i_ptr to_las(const void* p) {
  return (lv4i_ptr)(unsigned int)(uintptr_t)p;        // LDS offset = low 32 bits
}

// ---------------------------------------------------------------------------
// Pad + convert f32 -> bf16 (row-major, zero-padded).
// ---------------------------------------------------------------------------
__global__ __launch_bounds__(256) void pad_cvt(const float* __restrict__ src,
                                               __bf16* __restrict__ dst,
                                               int srows, int scols,
                                               int drows, int dcols) {
  int c = blockIdx.x * 256 + threadIdx.x;
  int r = blockIdx.y;
  if (c >= dcols || r >= drows) return;
  float v = (r < srows && c < scols) ? src[(size_t)r * scols + c] : 0.0f;
  dst[(size_t)r * dcols + c] = (__bf16)v;
}

// ---------------------------------------------------------------------------
// Transpose + convert:  dst[n][d] = (bf16) src[d][n], zero-padded.
// ---------------------------------------------------------------------------
__global__ __launch_bounds__(256) void cvt_transpose(const float* __restrict__ src,
                                                     __bf16* __restrict__ dst,
                                                     int srows, int scols,
                                                     int drows, int dcols) {
  __shared__ float t[32][33];
  const int tx = threadIdx.x & 31, ty = threadIdx.x >> 5;  // 32 x 8
  const int d0 = blockIdx.y * 32, n0 = blockIdx.x * 32;
#pragma unroll
  for (int i = ty; i < 32; i += 8) {
    int d = d0 + i, n = n0 + tx;
    t[i][tx] = (d < srows && n < scols) ? src[(size_t)d * scols + n] : 0.0f;
  }
  __syncthreads();
#pragma unroll
  for (int i = ty; i < 32; i += 8) {
    int n = n0 + i, d = d0 + tx;
    if (n < drows && d < dcols) dst[(size_t)n * dcols + d] = (__bf16)t[tx][i];
  }
}

// ---------------------------------------------------------------------------
// NT-GEMM, bf16 in / f32 accum via v_wmma_f32_16x16x32_bf16:
//   C[m,n] = sum_k A[m*lda + k] * B[n*ldb + k]
// Double-buffered LDS staging; tile k+1 streams global->LDS via
// global_load_async_to_lds_b128 (ASYNCcnt) while tile k runs on the WMMA
// pipe. Operand buffers are over-allocated to the 128-row tile grid so tile
// loads need no guards (exactly 4 async ops per wave per stage ->
// s_wait_asynccnt 4 gives exact pipelining; async loads complete in order).
//   OUT_MODE: 0 = bf16 store, 1 = f32 store, 2 = f32 with  Z + scale*acc
// ---------------------------------------------------------------------------
template <int OUT_MODE>
__global__ __launch_bounds__(256) void gemm_nt_bf16(
    const __bf16* __restrict__ A, const __bf16* __restrict__ B,
    void* __restrict__ Cout, const float* __restrict__ Zres,
    int Mstore, int N, int K, int lda, int ldb, int ldc, float scale) {
  __shared__ __bf16 As[2][128 * 40];  // [m][k], 40-elem row pitch (80B)
  __shared__ __bf16 Bs[2][128 * 40];  // [n][k]

  const int tid  = threadIdx.x;
  const int lane = tid & 31;
  const int wave = tid >> 5;  // 0..7
  const int wm   = wave >> 1; // 0..3 : 32-row wave tile
  const int wn   = wave & 1;  // 0..1 : 64-col wave tile
  const int m0   = blockIdx.y * 128;
  const int n0   = blockIdx.x * 128;
  const int half = lane >> 4;
  const int lmod = lane & 15;

  const int r  = tid >> 1;        // 0..127 staging row
  const int kq = (tid & 1) * 16;  // 0 / 16 staging k-quad

  auto stage = [&](int buf, int k0) {
    __bf16* as = &As[buf][0];
    __bf16* bs = &Bs[buf][0];
    const __bf16* ga = A + (size_t)(m0 + r) * lda + (k0 + kq);
    const __bf16* gb = B + (size_t)(n0 + r) * ldb + (k0 + kq);
#if HAVE_ASYNC
    __builtin_amdgcn_global_load_async_to_lds_b128(to_gas(ga),     to_las(&as[r * 40 + kq]),     0, 0);
    __builtin_amdgcn_global_load_async_to_lds_b128(to_gas(ga + 8), to_las(&as[r * 40 + kq + 8]), 0, 0);
    __builtin_amdgcn_global_load_async_to_lds_b128(to_gas(gb),     to_las(&bs[r * 40 + kq]),     0, 0);
    __builtin_amdgcn_global_load_async_to_lds_b128(to_gas(gb + 8), to_las(&bs[r * 40 + kq + 8]), 0, 0);
#else
    f32x4 a0 = *(const f32x4*)ga;
    f32x4 a1 = *(const f32x4*)(ga + 8);
    f32x4 b0 = *(const f32x4*)gb;
    f32x4 b1 = *(const f32x4*)(gb + 8);
    *(f32x4*)(&as[r * 40 + kq])     = a0;
    *(f32x4*)(&as[r * 40 + kq + 8]) = a1;
    *(f32x4*)(&bs[r * 40 + kq])     = b0;
    *(f32x4*)(&bs[r * 40 + kq + 8]) = b1;
#endif
  };

  v8f acc[2][4];
#pragma unroll
  for (int i = 0; i < 2; ++i)
#pragma unroll
    for (int j = 0; j < 4; ++j) acc[i][j] = (v8f)(0.0f);

  const int nIter = K / 32;
  stage(0, 0);
  for (int it = 0; it < nIter; ++it) {
    const int cur = it & 1;
    const bool hasNext = (it + 1) < nIter;
    if (hasNext) stage(cur ^ 1, (it + 1) * 32);
#if HAVE_ASYNC
    if (hasNext) WAIT_ASYNC(4);
    else         WAIT_ASYNC(0);
#endif
    __syncthreads();

    const __bf16* as = &As[cur][0];
    const __bf16* bs = &Bs[cur][0];
    // Fragments per CDNA5 16-bit WMMA VGPR layouts:
    // A: lane row M=lane%16; elems 0..7 -> K=c0..c0+7, 8..15 -> K=c0+16..+23,
//    c0 = 8*(lane>=16).  B: lane col N=lane%16; same K mapping as A.
    v16bf afrag[2];
    const int c0 = half * 8;
#pragma unroll
    for (int mt = 0; mt < 2; ++mt) {
      int row = wm * 32 + mt * 16 + lmod;
      v8bf x0 = *(const v8bf*)(&as[row * 40 + c0]);
      v8bf x1 = *(const v8bf*)(&as[row * 40 + c0 + 16]);
#pragma unroll
      for (int i = 0; i < 8; ++i) { afrag[mt][i] = x0[i]; afrag[mt][i + 8] = x1[i]; }
    }
    v16bf bfrag[4];
    const int kb = half * 8;
#pragma unroll
    for (int nt = 0; nt < 4; ++nt) {
      int col = wn * 64 + nt * 16 + lmod;
      v8bf y0 = *(const v8bf*)(&bs[col * 40 + kb]);
      v8bf y1 = *(const v8bf*)(&bs[col * 40 + kb + 16]);
#pragma unroll
      for (int i = 0; i < 8; ++i) { bfrag[nt][i] = y0[i]; bfrag[nt][i + 8] = y1[i]; }
    }

#pragma unroll
    for (int mt = 0; mt < 2; ++mt)
#pragma unroll
      for (int nt = 0; nt < 4; ++nt)
        acc[mt][nt] = __builtin_amdgcn_wmma_f32_16x16x32_bf16(
            false, afrag[mt], false, bfrag[nt], (short)0, acc[mt][nt], false, false);

    __syncthreads();
  }

  // ---- store: C/D layout row = v + 8*(lane>=16), col = lane%16 ----
#pragma unroll
  for (int mt = 0; mt < 2; ++mt)
#pragma unroll
    for (int nt = 0; nt < 4; ++nt)
#pragma unroll
      for (int v = 0; v < 8; ++v) {
        int gr = m0 + wm * 32 + mt * 16 + v + half * 8;
        int gc = n0 + wn * 64 + nt * 16 + lmod;
        if (gr < Mstore && gc < N) {
          float val = acc[mt][nt][v];
          size_t idx = (size_t)gr * ldc + gc;
          if (OUT_MODE == 0)      ((__bf16*)Cout)[idx] = (__bf16)val;
          else if (OUT_MODE == 1) ((float*)Cout)[idx]  = val;
          else                    ((float*)Cout)[idx]  = Zres[idx] + scale * val;
        }
      }
}

// ---------------------------------------------------------------------------
// Softmax row stats: mx[c] = max_m X[c,m]; inv[c] = 1/sum_m exp(X-mx).
// ---------------------------------------------------------------------------
__global__ __launch_bounds__(256) void softmax_stats(const float* __restrict__ X,
                                                     float* __restrict__ mx,
                                                     float* __restrict__ inv,
                                                     int n) {
  const int row = blockIdx.x;
  const float* xr = X + (size_t)row * n;
  __shared__ float red[8];
  const int tid = threadIdx.x, lane = tid & 31, wv = tid >> 5;

  float m = -3.4e38f;
  for (int c = tid; c < n; c += 256) m = fmaxf(m, xr[c]);
#pragma unroll
  for (int off = 16; off > 0; off >>= 1) m = fmaxf(m, __shfl_down(m, off, 32));
  if (lane == 0) red[wv] = m;
  __syncthreads();
  m = red[0];
#pragma unroll
  for (int i = 1; i < 8; ++i) m = fmaxf(m, red[i]);
  __syncthreads();

  float s = 0.0f;
  for (int c = tid; c < n; c += 256) s += expf(xr[c] - m);
#pragma unroll
  for (int off = 16; off > 0; off >>= 1) s += __shfl_down(s, off, 32);
  if (lane == 0) red[wv] = s;
  __syncthreads();
  if (tid == 0) {
    float tot = 0.0f;
#pragma unroll
    for (int i = 0; i < 8; ++i) tot += red[i];
    mx[row]  = m;
    inv[row] = 1.0f / tot;
  }
}

// ---------------------------------------------------------------------------
// At[m][c] = (bf16)( exp(X[c,m]-mx[c]) * inv[c] ), 64x64 LDS transpose.
// ---------------------------------------------------------------------------
__global__ __launch_bounds__(256) void exp_transpose(const float* __restrict__ X,
                                                     const float* __restrict__ mx,
                                                     const float* __restrict__ inv,
                                                     __bf16* __restrict__ At, int n) {
  __shared__ float t[64][65];
  const int tx = threadIdx.x & 63, ty = threadIdx.x >> 6;  // 64 x 4
  const int c0 = blockIdx.y * 64, m0 = blockIdx.x * 64;
#pragma unroll
  for (int i = ty; i < 64; i += 4) {
    int c = c0 + i;
    t[i][tx] = expf(X[(size_t)c * n + m0 + tx] - mx[c]) * inv[c];
  }
  __syncthreads();
#pragma unroll
  for (int i = ty; i < 64; i += 4) {
    int m = m0 + i;
    At[(size_t)m * n + c0 + tx] = (__bf16)t[tx][i];
  }
}

// ---------------------------------------------------------------------------
// Decay scan (PZ @ M, M[r,c]=lmbd^(r-c) lower-tri, last row/col zero).
//   PZ[d,r] = PZt[r*DPAD + d];  output PZM[d][r] (K-contiguous for GEMM 4).
// ---------------------------------------------------------------------------
__global__ __launch_bounds__(256) void scan_part(const __bf16* __restrict__ PZt,
                                                 float* __restrict__ Lbuf,
                                                 int dcols, int n) {
  const int d = blockIdx.x * 256 + threadIdx.x;
  const int c = blockIdx.y;  // chunk of 128 ctx cols
  if (d >= dcols) return;
  const int r0 = c * 128;
  float s = 0.0f;
  for (int k = 127; k >= 0; --k) {
    int rr = r0 + k;
    float v = (rr == n - 1) ? 0.0f : (float)PZt[(size_t)rr * dcols + d];
    s = v + LMBD * s;
  }
  Lbuf[(size_t)c * dcols + d] = s;
}

__global__ __launch_bounds__(256) void scan_combine(const float* __restrict__ Lbuf,
                                                    float* __restrict__ Ebuf,
                                                    int dcols) {
  const int d = blockIdx.x * 256 + threadIdx.x;
  if (d >= dcols) return;
  const float W = powf(LMBD, 128.0f);
  float e = 0.0f;
  for (int c = 31; c >= 0; --c) {
    Ebuf[(size_t)c * dcols + d] = e;
    e = Lbuf[(size_t)c * dcols + d] + W * e;
  }
}

__global__ __launch_bounds__(256) void scan_emit(const __bf16* __restrict__ PZt,
                                                 const float* __restrict__ Ebuf,
                                                 __bf16* __restrict__ PZM,
                                                 int dcols, int n) {
  const int d = blockIdx.x * 256 + threadIdx.x;
  const int c = blockIdx.y;
  if (d >= dcols) return;
  const int r0 = c * 128;
  float s = Ebuf[(size_t)c * dcols + d];
  for (int k = 127; k >= 0; --k) {
    int rr = r0 + k;
    float v = (rr == n - 1) ? 0.0f : (float)PZt[(size_t)rr * dcols + d];
    s = v + LMBD * s;
    PZM[(size_t)d * n + rr] = (__bf16)s;
  }
}

// ---------------------------------------------------------------------------
extern "C" void kernel_launch(void* const* d_in, const int* in_sizes, int n_in,
                              void* d_out, int out_size, void* d_ws, size_t ws_size,
                              hipStream_t stream) {
  const float* Z = (const float*)d_in[0];  // (1025, 4096)
  const float* P = (const float*)d_in[1];  // (1025, 1025)
  const float* Q = (const float*)d_in[2];  // (1025, 1025)
  // d_in[3] = M unused: its lmbd^(r-c) structure is applied as a decay scan.

  char* ws = (char*)d_ws;
  size_t off = 0;
  auto carve = [&](size_t bytes) -> char* {
    char* p = ws + off;
    off += (bytes + 255) & ~(size_t)255;
    return p;
  };
  // B/A operands read guard-free up to the 128-row tile grid -> pad rows.
  __bf16* Zt   = (__bf16*)carve((size_t)CTXX * DPAD * 2);     // Zt[n][d]
  __bf16* Qb   = (__bf16*)carve((size_t)DPAD128 * DPAD * 2);  // Qb[i][e]
  __bf16* Pb   = (__bf16*)carve((size_t)DPAD128 * DPAD * 2);
  __bf16* QZt  = (__bf16*)carve((size_t)CTXX * DPAD * 2);     // QZt[m][i]
  __bf16* PZt  = (__bf16*)carve((size_t)CTXX * DPAD * 2);
  __bf16* PZM  = (__bf16*)carve((size_t)DPAD128 * CTXX * 2);  // PZM[d][c]
  __bf16* At   = (__bf16*)carve((size_t)CTXX * CTXX * 2);     // At[m][c]
  float*  Xf   = (float*) carve((size_t)CTXX * CTXX * 4);     // X[n][m]
  float*  mxb  = (float*) carve((size_t)CTXX * 4);
  float*  invb = (float*) carve((size_t)CTXX * 4);
  float*  Lbuf = (float*) carve((size_t)32 * DPAD * 4);
  float*  Ebuf = (float*) carve((size_t)32 * DPAD * 4);

  // 1) pack/convert
  cvt_transpose<<<dim3(CTXX / 32, DPAD / 32), 256, 0, stream>>>(Z, Zt, DIMX, CTXX, CTXX, DPAD);
  pad_cvt<<<dim3((DPAD + 255) / 256, DPAD), 256, 0, stream>>>(Q, Qb, DIMX, DIMX, DPAD, DPAD);
  pad_cvt<<<dim3((DPAD + 255) / 256, DPAD), 256, 0, stream>>>(P, Pb, DIMX, DIMX, DPAD, DPAD);

  const dim3 gThin((DPAD + 127) / 128, CTXX / 128);  // 9 x 32  (N=1056)
  const dim3 gBig(CTXX / 128, CTXX / 128);           // 32 x 32
  const dim3 gOut(CTXX / 128, (DPAD + 127) / 128);   // 32 x 9

  // 2) QZt[m,i] = sum_e Zt[m,e]*Qb[i,e]   (== (Q@Z)^T, bf16)
  gemm_nt_bf16<0><<<gThin, 256, 0, stream>>>(Zt, Qb, QZt, nullptr,
      CTXX, DPAD, DPAD, DPAD, DPAD, DPAD, 0.0f);
  // 3) PZt = (P@Z)^T
  gemm_nt_bf16<0><<<gThin, 256, 0, stream>>>(Zt, Pb, PZt, nullptr,
      CTXX, DPAD, DPAD, DPAD, DPAD, DPAD, 0.0f);
  // 4) X[n,m] = sum_d Zt[n,d]*QZt[m,d]   (f32, lives in L2)
  gemm_nt_bf16<1><<<gBig, 256, 0, stream>>>(Zt, QZt, Xf, nullptr,
      CTXX, CTXX, DPAD, DPAD, DPAD, CTXX, 0.0f);
  // 5) softmax stats + fused exp/transpose -> At[m,c]
  softmax_stats<<<CTXX, 256, 0, stream>>>(Xf, mxb, invb, CTXX);
  exp_transpose<<<dim3(CTXX / 64, CTXX / 64), 256, 0, stream>>>(Xf, mxb, invb, At, CTXX);
  // 6) PZM[d,c] via decay scan over PZt
  scan_part<<<dim3((DPAD + 255) / 256, 32), 256, 0, stream>>>(PZt, Lbuf, DPAD, CTXX);
  scan_combine<<<dim3((DPAD + 255) / 256), 256, 0, stream>>>(Lbuf, Ebuf, DPAD);
  scan_emit<<<dim3((DPAD + 255) / 256, 32), 256, 0, stream>>>(PZt, Ebuf, PZM, DPAD, CTXX);
  // 7) out[d,m] = Z[d,m] + (1/N) * sum_c PZM[d,c]*At[m,c]
  gemm_nt_bf16<2><<<gOut, 256, 0, stream>>>(PZM, At, d_out, Z,
      DPAD, CTXX, CTXX, CTXX, CTXX, CTXX, 1.0f / 4095.0f);
}
